// GCN_25469156065546
// MI455X (gfx1250) — matrix operand order repaired and checked by reading, not compiled, not run on hardware
//
#include <hip/hip_runtime.h>

typedef __attribute__((ext_vector_type(16))) _Float16 v16h;
typedef __attribute__((ext_vector_type(8)))  _Float16 v8h;
typedef __attribute__((ext_vector_type(4)))  _Float16 v4h;
typedef __attribute__((ext_vector_type(8)))  float    v8f;

#define TM        16            // A row-tile per block
#define NOUT      64            // fixed output width (both GEMMs)
#define KCHUNK    128           // K staged per iteration
#define LDA_H     (KCHUNK + 8)  // padded A-tile row stride (halfs)
#define BFOOT     24            // padded per-(kgroup,n) segment (16 j-halfs + 8 pad)
#define NTHREADS  128           // 4 wave32s; wave w owns N-tile [16w, 16w+16)

// C[M x 64] = A[M x K] @ B[K x 64]  (+bias, tanh when EPILOGUE)
// A, B f32 row-major in global; converted to f16 while staging into LDS.
// B is staged in WMMA *fragment order* [kgroup][n][j] so each lane's 16
// B-halfs are contiguous -> 2x ds_load_b128 per fragment (no scalar u16s).
template <bool EPILOGUE>
__global__ __launch_bounds__(NTHREADS)
void gcn_gemm_f16_wmma(const float* __restrict__ A,
                       const float* __restrict__ B,
                       const float* __restrict__ bias,
                       float* __restrict__ C,
                       int K)
{
    __shared__ _Float16 aTile[TM * LDA_H];                    // 16 x 136 halfs
    __shared__ _Float16 bTile[(KCHUNK / 16) * NOUT * BFOOT];  // 8 x 64 x 24 halfs

    const int tid  = threadIdx.x;
    const int lane = tid & 31;
    const int wave = tid >> 5;
    const int n0   = wave * 16;                   // this wave's N-tile base
    const int m0   = blockIdx.x * TM;             // this block's row-tile base

    const int mrow = lane & 15;                   // M (A/C) / N (B) index in tile
    const int hi   = lane >> 4;                   // lane-group -> K sub-range

    v8f acc = {};                                 // 16x16 f32 accumulator

    for (int k0 = 0; k0 < K; k0 += KCHUNK) {
        // ---- stage A tile: 16 rows x 128 cols, coalesced float4 -> f16 b64 stores
        #pragma unroll
        for (int i = 0; i < (TM * KCHUNK / 4) / NTHREADS; ++i) {   // 4 iters
            int f  = tid + i * NTHREADS;
            int r  = f >> 5;                      // 32 float4 per 128-float row
            int c4 = f & 31;
            const float4 v = *(const float4*)(A + (size_t)(m0 + r) * K + k0 + c4 * 4);
            v4h h = { (_Float16)v.x, (_Float16)v.y, (_Float16)v.z, (_Float16)v.w };
            *(v4h*)&aTile[r * LDA_H + c4 * 4] = h;
        }

        // ---- stage B chunk in fragment order: each thread does 4 blocks of 4x4
        // bTile element for B[k][n]: [(k>>4)*64 + n] * BFOOT + (k & 15)
        #pragma unroll
        for (int i = 0; i < (KCHUNK * NOUT / 16) / NTHREADS; ++i) { // 4 iters
            int b  = tid + i * NTHREADS;          // block id 0..511
            int bc = b & 15;                      // col-block: n = bc*4 ..
            int br = b >> 4;                      // row-block: k = br*4 ..
            int kb = br * 4;
            const float* src = B + (size_t)(k0 + kb) * NOUT + bc * 4;
            const float4 r0 = *(const float4*)(src);
            const float4 r1 = *(const float4*)(src + NOUT);
            const float4 r2 = *(const float4*)(src + 2 * NOUT);
            const float4 r3 = *(const float4*)(src + 3 * NOUT);
            _Float16* dst = &bTile[((kb >> 4) * NOUT + bc * 4) * BFOOT + (kb & 15)];
            v4h c0 = { (_Float16)r0.x, (_Float16)r1.x, (_Float16)r2.x, (_Float16)r3.x };
            v4h c1 = { (_Float16)r0.y, (_Float16)r1.y, (_Float16)r2.y, (_Float16)r3.y };
            v4h c2 = { (_Float16)r0.z, (_Float16)r1.z, (_Float16)r2.z, (_Float16)r3.z };
            v4h c3 = { (_Float16)r0.w, (_Float16)r1.w, (_Float16)r2.w, (_Float16)r3.w };
            *(v4h*)(dst + 0 * BFOOT) = c0;
            *(v4h*)(dst + 1 * BFOOT) = c1;
            *(v4h*)(dst + 2 * BFOOT) = c2;
            *(v4h*)(dst + 3 * BFOOT) = c3;
        }

        // hint next chunk into cache while we compute (global_prefetch_b8)
        if (k0 + KCHUNK < K) {
            if (tid < TM)
                __builtin_prefetch(A + (size_t)(m0 + tid) * K + k0 + KCHUNK, 0, 0);
            __builtin_prefetch(B + (size_t)(k0 + KCHUNK + tid) * NOUT, 0, 0);
        }
        __syncthreads();

        // ---- 4x v_wmma_f32_16x16x32_f16 per wave over this K chunk
        #pragma unroll
        for (int kk = 0; kk < KCHUNK; kk += 32) {
            // A 16x32 fragment, this code's contiguous k order (not ISA table order): lane m=lane%16; lanes 0-15: K kk+0..15;
            //                 lanes 16-31:    K kk+16..31
            const _Float16* ap = &aTile[mrow * LDA_H + kk + hi * 16];
            v8h alo = *(const v8h*)(ap);          // 16B aligned
            v8h ahi = *(const v8h*)(ap + 8);
            v16h afrag = __builtin_shufflevector(alo, ahi,
                0,1,2,3,4,5,6,7,8,9,10,11,12,13,14,15);

            // B 32x16 fragment, same contiguous k order as A: n=lane%16; lanes 0-15: K kk+0..15; lanes 16-31: K kk+16..31
            const _Float16* bp = &bTile[(((kk >> 4) + hi) * NOUT + n0 + mrow) * BFOOT];
            v8h blo = *(const v8h*)(bp);          // 48-byte stride -> 16B aligned
            v8h bhi = *(const v8h*)(bp + 8);
            v16h bfrag = __builtin_shufflevector(blo, bhi,
                0,1,2,3,4,5,6,7,8,9,10,11,12,13,14,15);

            acc = __builtin_amdgcn_wmma_f32_16x16x32_f16(
                /*neg_a=*/false, afrag, /*neg_b=*/false, bfrag,
                /*c_mod=*/(short)0, acc, /*reuse_a=*/false, /*reuse_b=*/false);
        }
        __syncthreads();
    }

    // ---- epilogue + store (C/D: VGPR v -> M=v lanes 0-15, M=v+8 lanes 16-31)
    const int col = n0 + mrow;
    #pragma unroll
    for (int v = 0; v < 8; ++v) {
        int row = m0 + v + hi * 8;
        float val = acc[v];
        if constexpr (EPILOGUE) {
            val += bias[col];
            val = tanhf(val);
        }
        C[(size_t)row * NOUT + col] = val;
    }
}

extern "C" void kernel_launch(void* const* d_in, const int* in_sizes, int n_in,
                              void* d_out, int out_size, void* d_ws, size_t ws_size,
                              hipStream_t stream) {
    (void)in_sizes; (void)n_in; (void)out_size; (void)ws_size;
    const float* x      = (const float*)d_in[0];   // [16384, 128]
    const float* adj    = (const float*)d_in[1];   // [16384, 16384]
    const float* weight = (const float*)d_in[2];   // [128, 64]
    const float* bias   = (const float*)d_in[3];   // [64]
    float*       out    = (float*)d_out;           // [16384, 64]
    float*       support= (float*)d_ws;            // scratch: 16384*64 f32 = 4 MB

    const int N = 16384, INFEAT = 128;
    dim3 grid(N / TM), block(NTHREADS);

    // support = x @ W
    gcn_gemm_f16_wmma<false><<<grid, block, 0, stream>>>(x, weight, nullptr, support, INFEAT);
    // out = tanh(adj @ support + bias)
    gcn_gemm_f16_wmma<true ><<<grid, block, 0, stream>>>(adj, support, bias, out, N);
}
